// MultiModalAttention_6700148981814
// MI455X (gfx1250) — matrix operand order repaired and checked by reading, not compiled, not run on hardware
//
#include <hip/hip_runtime.h>

// ---------------------------------------------------------------------------
// MultiModalAttention for MI455X (gfx1250, wave32, WMMA bf16, async-to-LDS)
//   B=4, Sq=1024, Sk=2048, H=1024, NH=16, HD=64, scale = 1/8
// ---------------------------------------------------------------------------

typedef __bf16 bf16;
typedef __attribute__((ext_vector_type(16))) __bf16 v16bf;
typedef __attribute__((ext_vector_type(8)))  float  v8f;

union Frag16 { v16bf v; uint4 u[2]; };

__device__ __forceinline__ v8f wmma_bf16(v16bf a, v16bf b, v8f c) {
  // D = A(16x32 bf16) * B(32x16 bf16) + C(16x16 f32)
  return __builtin_amdgcn_wmma_f32_16x16x32_bf16(
      /*neg_a=*/false, a, /*neg_b=*/false, b,
      /*c_mod=*/(short)0, c, /*reuse_a=*/false, /*reuse_b=*/false);
}

// ---- VALU-only 16-lane butterfly max via DPP16 (no LDS traffic) ----
template <int CTRL>
__device__ __forceinline__ float dpp_f32(float x) {
  return __int_as_float(__builtin_amdgcn_update_dpp(
      0, __float_as_int(x), CTRL, 0xf, 0xf, true));
}
__device__ __forceinline__ float rmax16_dpp(float v) {
  v = fmaxf(v, dpp_f32<0xB1>(v));   // quad_perm [1,0,3,2]  (xor 1)
  v = fmaxf(v, dpp_f32<0x4E>(v));   // quad_perm [2,3,0,1]  (xor 2)
  v = fmaxf(v, dpp_f32<0x141>(v));  // row_half_mirror (join quads in 8-group)
  v = fmaxf(v, dpp_f32<0x140>(v));  // row_mirror (join 8-groups in 16-row)
  return v;
}

// ---- CDNA5 async global->LDS (ASYNCcnt); offset: advances LDS+mem addr ----
__device__ __forceinline__ void async_load_b128(uint32_t lds, uint32_t voff,
                                                uint64_t sbase) {
  asm volatile("global_load_async_to_lds_b128 %0, %1, %2"
               :: "v"(lds), "v"(voff), "s"(sbase) : "memory");
}
__device__ __forceinline__ void async_load_2xb128(uint32_t lds, uint32_t voff,
                                                  uint64_t sbase) {
  asm volatile("global_load_async_to_lds_b128 %0, %1, %2\n\t"
               "global_load_async_to_lds_b128 %0, %1, %2 offset:16"
               :: "v"(lds), "v"(voff), "s"(sbase) : "memory");
}
__device__ __forceinline__ void wait_async0() {
  asm volatile("s_wait_asynccnt 0" ::: "memory");
}

// ---------------------------------------------------------------------------
// Pre-pass: transpose-convert W (f32, [K=1024][N=1024]) -> Wt (bf16, [N][K])
// grid (32,32,4), block (32,8)
// ---------------------------------------------------------------------------
__global__ __launch_bounds__(256) void transpose_w(
    const float* __restrict__ W0, const float* __restrict__ W1,
    const float* __restrict__ W2, const float* __restrict__ W3,
    bf16* __restrict__ WtBase) {
  const float* Ws[4] = {W0, W1, W2, W3};
  const float* W = Ws[blockIdx.z];
  bf16* Wt = WtBase + (size_t)blockIdx.z * 1024 * 1024;
  __shared__ float tile[32][33];
  int n0 = blockIdx.x * 32, k0 = blockIdx.y * 32;
  int tx = threadIdx.x, ty = threadIdx.y;
#pragma unroll
  for (int i = 0; i < 4; ++i)
    tile[ty + 8 * i][tx] = W[(size_t)(k0 + ty + 8 * i) * 1024 + n0 + tx];
  __syncthreads();
#pragma unroll
  for (int i = 0; i < 4; ++i)
    Wt[(size_t)(n0 + ty + 8 * i) * 1024 + k0 + tx] = (bf16)tile[tx][ty + 8 * i];
}

// ---------------------------------------------------------------------------
// GEMM: C[M][1024] = A[M][1024] @ W + bias; double-buffered LDS, async B tile,
// one barrier per K-step. Tiles 128x128x32, 8 waves, wave = 64x32.
// EPI 0: Q -> bf16 [B,16,1024,64] *0.125 | EPI 1: K -> bf16 [B,16,2048,64]
// EPI 2: V -> bf16 [B,16,64,2048] (transposed) | EPI 3: O -> f32 [M][1024]
// ---------------------------------------------------------------------------
template <int EPI>
__global__ __launch_bounds__(256) void gemm_wmma(
    const void* __restrict__ Ap, const bf16* __restrict__ Bt,
    const float* __restrict__ bias, void* __restrict__ Outp) {
  constexpr bool A_F32 = (EPI != 3);
  const int m0 = blockIdx.y * 128;
  const int n0 = blockIdx.x * 128;
  const int t = threadIdx.x;
  const int lane = t & 31, wave = t >> 5;
  const int g = lane >> 4, lane16 = lane & 15;
  const int wm = wave >> 2, wn = wave & 3;

  __shared__ __align__(16) bf16 As[2][128][40];
  __shared__ __align__(16) bf16 Bs[2][128][40];

  const int arow = t >> 1, ahalf = t & 1;  // 2 threads/row, 16 elts each
  const uint32_t stage_voff = (uint32_t)((arow * 1024 + ahalf * 16) * 2);
  const uint64_t bbase = (uint64_t)(uintptr_t)(Bt + (size_t)n0 * 1024);

  float afs[16];
  auto loadA_f32 = [&](int k0) {
    const float* A = (const float*)Ap;
    const float4* src =
        (const float4*)(A + (size_t)(m0 + arow) * 1024 + k0 + ahalf * 16);
    *(float4*)(afs + 0) = src[0];
    *(float4*)(afs + 4) = src[1];
    *(float4*)(afs + 8) = src[2];
    *(float4*)(afs + 12) = src[3];
  };
  auto storeA_f32 = [&](int buf) {
    __align__(16) bf16 tmp[16];
#pragma unroll
    for (int i = 0; i < 16; ++i) tmp[i] = (bf16)afs[i];
    *(uint4*)&As[buf][arow][ahalf * 16] = ((const uint4*)tmp)[0];
    *(uint4*)(&As[buf][arow][ahalf * 16] + 8) = ((const uint4*)tmp)[1];
  };
  auto stage_async = [&](int k0, int buf) {
    async_load_2xb128((uint32_t)(uintptr_t)&Bs[buf][arow][ahalf * 16],
                      stage_voff, bbase + (uint64_t)k0 * 2);
    if constexpr (!A_F32) {
      const bf16* A = (const bf16*)Ap;
      async_load_2xb128((uint32_t)(uintptr_t)&As[buf][arow][ahalf * 16],
                        stage_voff,
                        (uint64_t)(uintptr_t)(A + (size_t)m0 * 1024 + k0));
    }
  };

  v8f acc[4][2];
#pragma unroll
  for (int i = 0; i < 4; ++i)
#pragma unroll
    for (int j = 0; j < 2; ++j)
#pragma unroll
      for (int e = 0; e < 8; ++e) acc[i][j][e] = 0.0f;

  // prologue: stage K-step 0 into buffer 0
  stage_async(0, 0);
  if constexpr (A_F32) { loadA_f32(0); storeA_f32(0); }
  wait_async0();

  for (int i = 0; i < 32; ++i) {
    __syncthreads();  // buffer (i&1) fully staged, buffer (i&1)^1 free
    const int buf = i & 1;
    const bool more = (i + 1) < 32;
    if (more) {
      stage_async((i + 1) * 32, buf ^ 1);
      if constexpr (A_F32) loadA_f32((i + 1) * 32);
    }
    // B fragments: lane = column n, K elems {8g..8g+7, 16+8g..16+8g+7}
    Frag16 b0, b1;
    {
      const bf16* r = &Bs[buf][wn * 32 + lane16][0];
      b0.u[0] = *(const uint4*)(r + 8 * g);
      b0.u[1] = *(const uint4*)(r + 16 + 8 * g);
    }
    {
      const bf16* r = &Bs[buf][wn * 32 + 16 + lane16][0];
      b1.u[0] = *(const uint4*)(r + 8 * g);
      b1.u[1] = *(const uint4*)(r + 16 + 8 * g);
    }
#pragma unroll
    for (int mt = 0; mt < 4; ++mt) {
      Frag16 a;  // A: lane = row m; K elems {8g..8g+7, 16+8g..16+8g+7}
      const bf16* r = &As[buf][wm * 64 + mt * 16 + lane16][0];
      a.u[0] = *(const uint4*)(r + 8 * g);
      a.u[1] = *(const uint4*)(r + 16 + 8 * g);
      acc[mt][0] = wmma_bf16(a.v, b0.v, acc[mt][0]);
      acc[mt][1] = wmma_bf16(a.v, b1.v, acc[mt][1]);
    }
    if (more) {
      if constexpr (A_F32) storeA_f32(buf ^ 1);
    }
    wait_async0();  // my async loads for i+1 landed before next barrier
  }

  // ---- epilogue: D layout lane = col n, VGPR r = row r + 8g ----
#pragma unroll
  for (int mt = 0; mt < 4; ++mt) {
#pragma unroll
    for (int nt = 0; nt < 2; ++nt) {
      int n = n0 + wn * 32 + nt * 16 + lane16;
      float bv = bias[n];
#pragma unroll
      for (int r = 0; r < 8; ++r) {
        int Mr = m0 + wm * 64 + mt * 16 + r + 8 * g;
        float val = acc[mt][nt][r] + bv;
        if constexpr (EPI == 0) {
          val *= 0.125f;  // fold 1/sqrt(64) into Q
          int b = Mr >> 10, s = Mr & 1023, h = n >> 6, d = n & 63;
          ((bf16*)Outp)[((size_t)(b * 16 + h) * 1024 + s) * 64 + d] = (bf16)val;
        } else if constexpr (EPI == 1) {
          int b = Mr >> 11, s = Mr & 2047, h = n >> 6, d = n & 63;
          ((bf16*)Outp)[((size_t)(b * 16 + h) * 2048 + s) * 64 + d] = (bf16)val;
        } else if constexpr (EPI == 2) {
          int b = Mr >> 11, s = Mr & 2047, h = n >> 6, d = n & 63;
          ((bf16*)Outp)[((size_t)(b * 16 + h) * 64 + d) * 2048 + s] = (bf16)val;
        } else {
          ((float*)Outp)[(size_t)Mr * 1024 + n] = val;
        }
      }
    }
  }
}

// ---------------------------------------------------------------------------
// Flash attention: grid (8 qtiles, 16 heads, 4 batch), 256 threads (8 waves).
// Double-buffered async K/V staging (1 barrier/chunk). Online softmax:
// row-max via DPP butterfly, row-sum via ones-column WMMA (l lives in o[4]).
// ---------------------------------------------------------------------------
__global__ __launch_bounds__(256) void attn_flash(
    const bf16* __restrict__ Qbf, const bf16* __restrict__ Kbf,
    const bf16* __restrict__ Vt, const int* __restrict__ mask,
    bf16* __restrict__ Obf) {
  const int qt = blockIdx.x, h = blockIdx.y, b = blockIdx.z;
  const int t = threadIdx.x;
  const int lane = t & 31, wave = t >> 5;
  const int g = lane >> 4, lane16 = lane & 15;
  const int qbase = qt * 128 + wave * 16;
  const size_t bh = (size_t)(b * 16 + h);

  __shared__ __align__(16) bf16 Ks[2][32][72];     // [s_local][d]
  __shared__ __align__(16) bf16 Vs[2][64][40];     // [d][s_local]
  __shared__ __align__(16) bf16 Ps[8][16][40];     // per-wave P scratch
  __shared__ float biasS[2][32];

  // Q fragments (HD=64 -> two K=32 fragments), registers for entire kernel
  Frag16 q0, q1;
  {
    const bf16* qrow = Qbf + (bh * 1024 + qbase + lane16) * 64;
    q0.u[0] = *(const uint4*)(qrow + 8 * g);
    q0.u[1] = *(const uint4*)(qrow + 16 + 8 * g);
    q1.u[0] = *(const uint4*)(qrow + 32 + 8 * g);
    q1.u[1] = *(const uint4*)(qrow + 48 + 8 * g);
  }
  // ones B-fragment: o[4] accumulates row-sums of P (the softmax denominator)
  Frag16 onesf;
#pragma unroll
  for (int e = 0; e < 16; ++e) onesf.v[e] = (bf16)1.0f;

  float mrun[8];
  v8f o[5];
#pragma unroll
  for (int r = 0; r < 8; ++r) mrun[r] = -1e30f;
#pragma unroll
  for (int dt = 0; dt < 5; ++dt)
#pragma unroll
    for (int r = 0; r < 8; ++r) o[dt][r] = 0.0f;

  const int krow = t >> 3, kseg = t & 7;  // K: 32 rows x 8 segs of 8 bf16
  const int vrow = t >> 2, vseg = t & 3;  // V: 64 rows x 4 segs of 8 bf16
  const uint64_t kbase = (uint64_t)(uintptr_t)(Kbf + bh * 2048 * 64);
  const uint64_t vbase = (uint64_t)(uintptr_t)(Vt + bh * 64 * 2048);
  const uint32_t kvoff = (uint32_t)((krow * 64 + kseg * 8) * 2);
  const uint32_t vvoff = (uint32_t)((vrow * 2048 + vseg * 8) * 2);

  auto stage = [&](int c, int buf) {
    const int sbase = c * 32;
    async_load_b128((uint32_t)(uintptr_t)&Ks[buf][krow][kseg * 8], kvoff,
                    kbase + (uint64_t)sbase * 128);
    async_load_b128((uint32_t)(uintptr_t)&Vs[buf][vrow][vseg * 8], vvoff,
                    vbase + (uint64_t)sbase * 2);
    if (t < 32) biasS[buf][t] = mask[b * 2048 + sbase + t] ? 0.0f : -1e30f;
  };

  stage(0, 0);
  wait_async0();

  for (int c = 0; c < 64; ++c) {
    __syncthreads();  // buffer (c&1) staged by all waves; other buffer free
    const int buf = c & 1;
    if (c + 1 < 64) stage(c + 1, buf ^ 1);

    // ---- scores: S(16x32) = Q(16x64) . K^T, two 16x16 n-tiles ----
    v8f s0, s1;
#pragma unroll
    for (int e = 0; e < 8; ++e) { s0[e] = 0.0f; s1[e] = 0.0f; }
    {
      Frag16 kf;
      const bf16* r0 = &Ks[buf][lane16][0];
      kf.u[0] = *(const uint4*)(r0 + 8 * g);
      kf.u[1] = *(const uint4*)(r0 + 16 + 8 * g);
      s0 = wmma_bf16(q0.v, kf.v, s0);
      kf.u[0] = *(const uint4*)(r0 + 32 + 8 * g);
      kf.u[1] = *(const uint4*)(r0 + 48 + 8 * g);
      s0 = wmma_bf16(q1.v, kf.v, s0);
      const bf16* r1 = &Ks[buf][16 + lane16][0];
      kf.u[0] = *(const uint4*)(r1 + 8 * g);
      kf.u[1] = *(const uint4*)(r1 + 16 + 8 * g);
      s1 = wmma_bf16(q0.v, kf.v, s1);
      kf.u[0] = *(const uint4*)(r1 + 32 + 8 * g);
      kf.u[1] = *(const uint4*)(r1 + 48 + 8 * g);
      s1 = wmma_bf16(q1.v, kf.v, s1);
    }

    // ---- online softmax (row r+8g spread over 16 lanes; DPP reductions) ----
    const float bias0 = biasS[buf][lane16];
    const float bias1 = biasS[buf][16 + lane16];
    float alpha[8];
#pragma unroll
    for (int r = 0; r < 8; ++r) {
      float v0 = s0[r] + bias0;
      float v1 = s1[r] + bias1;
      float mnew = fmaxf(mrun[r], rmax16_dpp(fmaxf(v0, v1)));
      alpha[r] = __expf(mrun[r] - mnew);
      mrun[r] = mnew;
      Ps[wave][r + 8 * g][lane16] = (bf16)__expf(v0 - mnew);
      Ps[wave][r + 8 * g][16 + lane16] = (bf16)__expf(v1 - mnew);
    }
#pragma unroll
    for (int dt = 0; dt < 5; ++dt)
#pragma unroll
      for (int r = 0; r < 8; ++r) o[dt][r] *= alpha[r];

    // wave-private P relayout: LDS is in-order within a wave; just stop the
    // compiler from moving the loads above the stores.
    __builtin_amdgcn_wave_barrier();

    Frag16 pf;
    {
      const bf16* pr = &Ps[wave][lane16][0];
      pf.u[0] = *(const uint4*)(pr + 8 * g);
      pf.u[1] = *(const uint4*)(pr + 16 + 8 * g);
    }
#pragma unroll
    for (int dt = 0; dt < 4; ++dt) {
      Frag16 vf;
      const bf16* vr = &Vs[buf][dt * 16 + lane16][0];
      vf.u[0] = *(const uint4*)(vr + 8 * g);
      vf.u[1] = *(const uint4*)(vr + 16 + 8 * g);
      o[dt] = wmma_bf16(pf.v, vf.v, o[dt]);
    }
    o[4] = wmma_bf16(pf.v, onesf.v, o[4]);  // denominator accumulation

    wait_async0();  // next chunk landed in other buffer before next barrier
  }

  // ---- normalize and store as [B, Sq, H] bf16 (heads concatenated) ----
#pragma unroll
  for (int r = 0; r < 8; ++r) {
    float inv = 1.0f / o[4][r];
    int q = qbase + r + 8 * g;
#pragma unroll
    for (int dt = 0; dt < 4; ++dt) {
      Obf[((size_t)b * 1024 + q) * 1024 + h * 64 + dt * 16 + lane16] =
          (bf16)(o[dt][r] * inv);
    }
  }
}

// ---------------------------------------------------------------------------
// Launch
// ---------------------------------------------------------------------------
extern "C" void kernel_launch(void* const* d_in, const int* in_sizes, int n_in,
                              void* d_out, int out_size, void* d_ws,
                              size_t ws_size, hipStream_t stream) {
  (void)in_sizes; (void)n_in; (void)out_size; (void)ws_size;
  const float* query = (const float*)d_in[0];
  const float* key_  = (const float*)d_in[1];
  const float* value = (const float*)d_in[2];
  const int*   mask  = (const int*)d_in[3];
  const float* Wq = (const float*)d_in[4];
  const float* bq = (const float*)d_in[5];
  const float* Wk = (const float*)d_in[6];
  const float* bk = (const float*)d_in[7];
  const float* Wv = (const float*)d_in[8];
  const float* bv = (const float*)d_in[9];
  const float* Wo = (const float*)d_in[10];
  const float* bo = (const float*)d_in[11];

  char* ws = (char*)d_ws;
  const size_t MB = 1u << 20;
  bf16* Wt  = (bf16*)(ws);              // 4 x 2MB  (W^T, bf16, [N][K])
  bf16* Qbf = (bf16*)(ws + 8 * MB);     // 8MB  [4,16,1024,64]
  bf16* Kbf = (bf16*)(ws + 16 * MB);    // 16MB [4,16,2048,64]
  bf16* Vtb = (bf16*)(ws + 32 * MB);    // 16MB [4,16,64,2048]
  bf16* Obf = (bf16*)(ws + 48 * MB);    // 8MB  [4,1024,1024]

  transpose_w<<<dim3(32, 32, 4), dim3(32, 8), 0, stream>>>(Wq, Wk, Wv, Wo, Wt);

  gemm_wmma<0><<<dim3(8, 32), 256, 0, stream>>>(query, Wt + 0 * 1048576, bq, Qbf);
  gemm_wmma<1><<<dim3(8, 64), 256, 0, stream>>>(key_,  Wt + 1 * 1048576, bk, Kbf);
  gemm_wmma<2><<<dim3(8, 64), 256, 0, stream>>>(value, Wt + 2 * 1048576, bv, Vtb);

  attn_flash<<<dim3(8, 16, 4), 256, 0, stream>>>(Qbf, Kbf, Vtb, mask, Obf);

  gemm_wmma<3><<<dim3(8, 32), 256, 0, stream>>>(Obf, Wt + 3 * 1048576, bo, d_out);
}
